// Attention_74835510166020
// MI455X (gfx1250) — matrix operand order repaired and checked by reading, not compiled, not run on hardware
//
#include <hip/hip_runtime.h>
#include <hip/hip_bf16.h>

typedef __attribute__((ext_vector_type(16))) __bf16 v16bf;
typedef __attribute__((ext_vector_type(8)))  float  v8f;

#define WMMA_BF16(a, b, c) \
  __builtin_amdgcn_wmma_f32_16x16x32_bf16(false, (a), false, (b), (short)0, (c), false, false)

// ---------------------------------------------------------------------------
// CDNA5 async global->LDS copy (ASYNCcnt path, CDNA5 ISA).
// Per-lane: LDS[vdst] <= MEM[saddr + vaddr], 16 bytes. GVS addressing mode.
// ---------------------------------------------------------------------------
__device__ __forceinline__ void async_tile_load(unsigned lds_base,
                                                const __bf16* gbase,
                                                int bytes, int tid) {
  unsigned long long g = (unsigned long long)(uintptr_t)gbase;
  for (int off = tid * 16; off < bytes; off += 256 * 16) {
    unsigned lds = lds_base + (unsigned)off;
    asm volatile("global_load_async_to_lds_b128 %0, %1, %2"
                 : : "v"(lds), "v"(off), "s"(g) : "memory");
  }
}

__device__ __forceinline__ void wait_async0() {
  asm volatile("s_wait_asynccnt 0" ::: "memory");
}

// ---------------------------------------------------------------------------
// WMMA fragment loaders (wave32; contiguous k = 16*half + s, same order in A and B)
// ---------------------------------------------------------------------------

// A-matrix 16x32 bf16 frag. base -> element (row0, k0), row-major, ld elements.
__device__ __forceinline__ v16bf load_A_frag(const __bf16* base, int ld, int lane) {
  const int half = lane >> 4, m = lane & 15;
  const __bf16* rowp = base + m * ld;
  union { v16bf v; unsigned u[8]; } r;
#pragma unroll
  for (int i = 0; i < 8; ++i) {
    int kb = half * 16 + 2 * i;  // even -> dword aligned
    r.u[i] = *(const unsigned*)(rowp + kb);
  }
  return r.v;
}

// B-matrix 32x16 bf16 frag from a *transposed* LDS tile: element (k,n) at
// base[n*ld + k] (k contiguous). Used for Q*K^T where K tile is [key][d].
__device__ __forceinline__ v16bf load_Bt_frag(const __bf16* base, int ld, int lane) {
  const int half = lane >> 4, n = lane & 15;
  const __bf16* colp = base + n * ld + half * 16;
  union { v16bf v; unsigned u[8]; } r;
#pragma unroll
  for (int i = 0; i < 8; ++i) r.u[i] = *(const unsigned*)(colp + 2 * i);
  return r.v;
}

// B-matrix 32x16 bf16 frag from a k-pair-packed tile:
// element (k,n) stored at base[(k>>1)*ldp + 2n + (k&1)].
__device__ __forceinline__ v16bf load_Bp_frag(const __bf16* base, int ldp, int lane) {
  const int half = lane >> 4, n = lane & 15;
  const __bf16* p = base + half * 8 * ldp + 2 * n;
  union { v16bf v; unsigned u[8]; } r;
#pragma unroll
  for (int i = 0; i < 8; ++i) r.u[i] = *(const unsigned*)(p + i * ldp);
  return r.v;
}

// Workspace tile geometry (bf16 elements)
#define QK_TILE 9216    // 128 rows x 72 (64 + LDS pad baked in)
#define VP_TILE 8704    // 64 packed rows x 136

// ---------------------------------------------------------------------------
// Kernel 1: QKV GEMM  [8192x1024] x [1024x3072] + bias, head-split scatter.
// Q,K -> bf16 ws tiles (padded ld 72); V -> bf16 ws k-pair-packed tiles;
// K,V -> fp32 present region of d_out [B,H,S,D].
// ---------------------------------------------------------------------------
__global__ __launch_bounds__(256) void qkv_gemm_kernel(
    const float* __restrict__ x, const float* __restrict__ Wqkv,
    const float* __restrict__ bqkv,
    __bf16* __restrict__ qbf, __bf16* __restrict__ kbf, __bf16* __restrict__ vp,
    float* __restrict__ pk, float* __restrict__ pv) {
  __shared__ __bf16 As[128][40];    // [m][k], k-pad 8
  __shared__ __bf16 Bsp[16][264];   // packed [k>>1][2n + (k&1)]
  const int tid = threadIdx.x;
  const int lane = tid & 31, wave = tid >> 5;
  const int wm = wave >> 1, wn = wave & 1;
  const int m0 = blockIdx.y * 128;
  const int n0 = blockIdx.x * 128;

  v8f acc[2][4] = {};

  for (int kt = 0; kt < 1024; kt += 32) {
#pragma unroll
    for (int i = 0; i < 16; ++i) {                    // A tile 128x32 (cvt)
      int e = tid + 256 * i, r = e >> 5, c = e & 31;
      As[r][c] = (__bf16)x[(size_t)(m0 + r) * 1024 + kt + c];
    }
#pragma unroll
    for (int i = 0; i < 16; ++i) {                    // B tile 32x128, packed
      int e = tid + 256 * i, r = e >> 7, c = e & 127;
      Bsp[r >> 1][2 * c + (r & 1)] = (__bf16)Wqkv[(size_t)(kt + r) * 3072 + n0 + c];
    }
    __syncthreads();
    v16bf a0 = load_A_frag(&As[wm * 32][0], 40, lane);
    v16bf a1 = load_A_frag(&As[wm * 32 + 16][0], 40, lane);
#pragma unroll
    for (int j = 0; j < 4; ++j) {
      v16bf b = load_Bp_frag(&Bsp[0][2 * (wn * 64 + j * 16)], 264, lane);
      acc[0][j] = WMMA_BF16(a0, b, acc[0][j]);
      acc[1][j] = WMMA_BF16(a1, b, acc[1][j]);
    }
    __syncthreads();
  }

  const int half = lane >> 4, nc = lane & 15;
#pragma unroll
  for (int j = 0; j < 4; ++j) {
    int n = n0 + wn * 64 + j * 16 + nc;
    float bias = bqkv[n];
    int sec = n >> 10;               // 0=q 1=k 2=v
    int c = n & 1023, h = c >> 6, d = c & 63;
#pragma unroll
    for (int i = 0; i < 2; ++i) {
#pragma unroll
      for (int e = 0; e < 8; ++e) {
        int m = m0 + wm * 32 + i * 16 + e + half * 8;
        int b = m >> 10, s = m & 1023;
        int bh = b * 16 + h;
        int st = s >> 7, s7 = s & 127;
        float val = acc[i][j][e] + bias;
        if (sec == 0) {
          qbf[((size_t)bh * 8 + st) * QK_TILE + s7 * 72 + d] = (__bf16)val;
        } else if (sec == 1) {
          size_t idx = ((size_t)bh * 1024 + s) * 64 + d;
          pk[idx] = val;
          kbf[((size_t)bh * 8 + st) * QK_TILE + s7 * 72 + d] = (__bf16)val;
        } else {
          size_t idx = ((size_t)bh * 1024 + s) * 64 + d;
          pv[idx] = val;
          vp[((size_t)bh * 8 + st) * VP_TILE + (s7 >> 1) * 136 + 2 * d + (s7 & 1)] =
              (__bf16)val;
        }
      }
    }
  }
}

// ---------------------------------------------------------------------------
// Kernel 2: causal flash attention. One block per (b*H+h, 128-query tile).
// Q/K/V tiles arrive in LDS via global_load_async_to_lds_b128 (no conversion).
// ---------------------------------------------------------------------------
__global__ __launch_bounds__(256) void attn_kernel(
    const __bf16* __restrict__ qbf, const __bf16* __restrict__ kbf,
    const __bf16* __restrict__ vp, __bf16* __restrict__ attnbf) {
  __shared__ __bf16 Qs[128][72];    // [q][d]
  __shared__ __bf16 Ks[128][72];    // [key][d]  (transposed-B access)
  __shared__ __bf16 Vsp[64][136];   // packed [key>>1][2d + (key&1)]
  __shared__ __bf16 Ps[128][136];   // [q][key] probabilities (wave-private rows)

  const int tid = threadIdx.x, lane = tid & 31, wave = tid >> 5;
  const int half = lane >> 4, nc = lane & 15;
  const int qt = blockIdx.x, bh = blockIdx.y;

  const unsigned ldsQ = (unsigned)(uintptr_t)&Qs[0][0];
  const unsigned ldsK = (unsigned)(uintptr_t)&Ks[0][0];
  const unsigned ldsV = (unsigned)(uintptr_t)&Vsp[0][0];

  // async load Q tile (18432 B)
  async_tile_load(ldsQ, qbf + ((size_t)bh * 8 + qt) * QK_TILE, QK_TILE * 2, tid);
  wait_async0();
  __syncthreads();
  const v16bf qa0 = load_A_frag(&Qs[wave * 16][0], 72, lane);
  const v16bf qa1 = load_A_frag(&Qs[wave * 16][32], 72, lane);

  v8f o[4] = {};
  float rmax[8], rsum[8];
#pragma unroll
  for (int e = 0; e < 8; ++e) { rmax[e] = -3.0e38f; rsum[e] = 0.0f; }

  for (int j = 0; j <= qt; ++j) {
    async_tile_load(ldsK, kbf + ((size_t)bh * 8 + j) * QK_TILE, QK_TILE * 2, tid);
    async_tile_load(ldsV, vp + ((size_t)bh * 8 + j) * VP_TILE, VP_TILE * 2, tid);
    wait_async0();
    __syncthreads();

    // S = Q K^T : wave covers 16 q-rows x 128 keys (8 tiles, 2 k-steps each)
    v8f sacc[8] = {};
#pragma unroll
    for (int t = 0; t < 8; ++t) {
      v16bf b0 = load_Bt_frag(&Ks[t * 16][0], 72, lane);
      v16bf b1 = load_Bt_frag(&Ks[t * 16][32], 72, lane);
      sacc[t] = WMMA_BF16(qa0, b0, sacc[t]);
      sacc[t] = WMMA_BF16(qa1, b1, sacc[t]);
    }

    // Online softmax (rows e + 8*half of this wave's 16-row band)
#pragma unroll
    for (int e = 0; e < 8; ++e) {
      int qr = qt * 128 + wave * 16 + e + half * 8;
      float mm = -3.0e38f;
#pragma unroll
      for (int t = 0; t < 8; ++t) {
        float sv = sacc[t][e] * 0.125f;           // 1/sqrt(64)
        int kc = j * 128 + t * 16 + nc;
        if (kc > qr) sv = -10000.0f;              // reference causal mask value
        sacc[t][e] = sv;
        mm = fmaxf(mm, sv);
      }
#pragma unroll
      for (int off = 1; off < 16; off <<= 1)
        mm = fmaxf(mm, __shfl_xor(mm, off, 32));
      float mnew = fmaxf(rmax[e], mm);
      float sc = __expf(rmax[e] - mnew);
      rmax[e] = mnew;
      float ps = 0.0f;
      int prow = wave * 16 + e + half * 8;
#pragma unroll
      for (int t = 0; t < 8; ++t) {
        float p = __expf(sacc[t][e] - mnew);
        Ps[prow][t * 16 + nc] = (__bf16)p;        // wave-private rows: no barrier
        ps += p;
      }
#pragma unroll
      for (int off = 1; off < 16; off <<= 1)
        ps += __shfl_xor(ps, off, 32);
      rsum[e] = rsum[e] * sc + ps;
#pragma unroll
      for (int jj = 0; jj < 4; ++jj) o[jj][e] *= sc;
    }

    // O += P V : 16 rows x 64 cols, K=128 keys in 4 chunks of 32
    v16bf pa[4];
#pragma unroll
    for (int ks = 0; ks < 4; ++ks)
      pa[ks] = load_A_frag(&Ps[wave * 16][ks * 32], 136, lane);
#pragma unroll
    for (int jj = 0; jj < 4; ++jj) {
#pragma unroll
      for (int ks = 0; ks < 4; ++ks) {
        v16bf vb = load_Bp_frag(&Vsp[ks * 16][2 * (jj * 16)], 136, lane);
        o[jj] = WMMA_BF16(pa[ks], vb, o[jj]);
      }
    }
    __syncthreads();   // protect Ks/Vsp before next tile's async writes
  }

  // normalize and write merged-head bf16 [B,S,NX] for proj GEMM
  const int b = bh >> 4, h = bh & 15;
#pragma unroll
  for (int e = 0; e < 8; ++e) {
    float inv = 1.0f / rsum[e];
    int s = qt * 128 + wave * 16 + e + half * 8;
    size_t rowbase = ((size_t)b * 1024 + s) * 1024 + h * 64;
#pragma unroll
    for (int jj = 0; jj < 4; ++jj)
      attnbf[rowbase + jj * 16 + nc] = (__bf16)(o[jj][e] * inv);
  }
}

// ---------------------------------------------------------------------------
// Kernel 3: projection GEMM [8192x1024] x [1024x1024] + bias -> fp32 a.
// A tile (already bf16) staged with per-lane async row gathers.
// ---------------------------------------------------------------------------
__global__ __launch_bounds__(256) void proj_gemm_kernel(
    const __bf16* __restrict__ attnbf, const float* __restrict__ Wproj,
    const float* __restrict__ bproj, float* __restrict__ out) {
  __shared__ __bf16 As[128][40];    // ld 40 bf16 = 80 B
  __shared__ __bf16 Bsp[16][264];
  const int tid = threadIdx.x;
  const int lane = tid & 31, wave = tid >> 5;
  const int wm = wave >> 1, wn = wave & 1;
  const int m0 = blockIdx.y * 128;
  const int n0 = blockIdx.x * 128;

  const unsigned ldsA = (unsigned)(uintptr_t)&As[0][0];
  const unsigned long long gA = (unsigned long long)(uintptr_t)attnbf;

  v8f acc[2][4] = {};

  for (int kt = 0; kt < 1024; kt += 32) {
    // A tile 128x32 bf16: 128 rows x 4 chunks of 16B, per-lane async gather
#pragma unroll
    for (int i = 0; i < 2; ++i) {
      int e = tid + 256 * i, r = e >> 2, c = e & 3;
      int goff = (m0 + r) * 2048 + kt * 2 + c * 16;
      unsigned lds = ldsA + (unsigned)(r * 80 + c * 16);
      asm volatile("global_load_async_to_lds_b128 %0, %1, %2"
                   : : "v"(lds), "v"(goff), "s"(gA) : "memory");
    }
#pragma unroll
    for (int i = 0; i < 16; ++i) {                    // B tile (cvt), packed
      int e = tid + 256 * i, r = e >> 7, c = e & 127;
      Bsp[r >> 1][2 * c + (r & 1)] = (__bf16)Wproj[(size_t)(kt + r) * 1024 + n0 + c];
    }
    wait_async0();
    __syncthreads();
    v16bf a0 = load_A_frag(&As[wm * 32][0], 40, lane);
    v16bf a1 = load_A_frag(&As[wm * 32 + 16][0], 40, lane);
#pragma unroll
    for (int j = 0; j < 4; ++j) {
      v16bf b = load_Bp_frag(&Bsp[0][2 * (wn * 64 + j * 16)], 264, lane);
      acc[0][j] = WMMA_BF16(a0, b, acc[0][j]);
      acc[1][j] = WMMA_BF16(a1, b, acc[1][j]);
    }
    __syncthreads();
  }

  const int half = lane >> 4, nc = lane & 15;
#pragma unroll
  for (int j = 0; j < 4; ++j) {
    int n = n0 + wn * 64 + j * 16 + nc;
    float bias = bproj[n];
#pragma unroll
    for (int i = 0; i < 2; ++i) {
#pragma unroll
      for (int e = 0; e < 8; ++e) {
        int m = m0 + wm * 32 + i * 16 + e + half * 8;
        out[(size_t)m * 1024 + n] = acc[i][j][e] + bias;
      }
    }
  }
}

// ---------------------------------------------------------------------------
extern "C" void kernel_launch(void* const* d_in, const int* in_sizes, int n_in,
                              void* d_out, int out_size, void* d_ws, size_t ws_size,
                              hipStream_t stream) {
  const float* x     = (const float*)d_in[0];
  const float* Wqkv  = (const float*)d_in[1];
  const float* bqkv  = (const float*)d_in[2];
  const float* Wproj = (const float*)d_in[3];
  const float* bproj = (const float*)d_in[4];

  float* out = (float*)d_out;
  const size_t A_ELEMS = (size_t)8 * 1024 * 1024;    // B*S*NX
  float* pk = out + A_ELEMS;                         // present[0] = key   [B,H,S,D]
  float* pv = pk + A_ELEMS;                          // present[1] = value [B,H,S,D]

  const size_t NTILES = 128 * 8;                     // (B*H) * (S/128)
  __bf16* qbf    = (__bf16*)d_ws;                    // NTILES * QK_TILE
  __bf16* kbf    = qbf + NTILES * QK_TILE;
  __bf16* vp     = kbf + NTILES * QK_TILE;           // NTILES * VP_TILE
  __bf16* attnbf = vp + NTILES * VP_TILE;            // A_ELEMS bf16

  qkv_gemm_kernel<<<dim3(24, 64), 256, 0, stream>>>(x, Wqkv, bqkv, qbf, kbf, vp, pk, pv);
  attn_kernel<<<dim3(8, 128), 256, 0, stream>>>(qbf, kbf, vp, attnbf);
  proj_gemm_kernel<<<dim3(8, 64), 256, 0, stream>>>(attnbf, Wproj, bproj, out);
}
